// SelfAttention_85555748537144
// MI455X (gfx1250) — compile-verified, matrix operand order checked by reading, not run on hardware
//
#include <hip/hip_runtime.h>
#include <hip/hip_bf16.h>

// ---------------- CDNA5 WMMA types ----------------
typedef __bf16 v16bf __attribute__((ext_vector_type(16)));
typedef float  v8f   __attribute__((ext_vector_type(8)));

#define T_LEN   1024
#define MEM_LEN 1024
#define S_LEN   2048   // t + mem
#define D_MODEL 1024
#define HEADS   16
#define DIM_H   64
#define SCALE   0.125f // 64^-0.5

static __device__ __forceinline__ unsigned short f2bf(float f) {
    unsigned int u = __float_as_uint(f);
    unsigned int r = u + 0x7FFFu + ((u >> 16) & 1u);   // round-to-nearest-even
    return (unsigned short)(r >> 16);
}

union Frag16 {
    v16bf v;
    unsigned short u[16];
    uint4 q[2];
};

// Load a 16x32 bf16 A/B fragment from row-major bf16 storage.
// p: per-lane row base (already includes row*stride + k-chunk base), kh = (lane>=16)?8:0
static __device__ __forceinline__ v16bf ldfrag_bf16(const unsigned short* p, int kh) {
    Frag16 f;
    f.q[0] = *(const uint4*)(p + kh);        // k = kh .. kh+7
    f.q[1] = *(const uint4*)(p + kh + 16);   // k = 16+kh .. 16+kh+7
    return f.v;
}

static __device__ __forceinline__ v8f wmma_bf16(v16bf a, v16bf b, v8f c) {
    return __builtin_amdgcn_wmma_f32_16x16x32_bf16(false, a, false, b, (short)0, c, false, false);
}

// ---------------- pass 0a: elementwise fp32 -> bf16 convert ----------------
__global__ __launch_bounds__(256) void convert_bf16_kernel(const float* __restrict__ src,
                                                           unsigned short* __restrict__ dst,
                                                           long long total) {
    long long i = (long long)blockIdx.x * blockDim.x + threadIdx.x;
    if (i < total) dst[i] = f2bf(src[i]);
}

// ---------------- pass 0b: weight transpose + bf16 convert ----------------
// src: fp32 [K][N] row-major, dst: bf16 [N][K] row-major (K contiguous for B-fragments)
__global__ __launch_bounds__(256) void transpose_bf16_kernel(const float* __restrict__ src,
                                                             unsigned short* __restrict__ dst,
                                                             int K, int N) {
    long long i = (long long)blockIdx.x * blockDim.x + threadIdx.x;
    long long total = (long long)K * N;
    if (i < total) {
        int n = (int)(i / K);
        int k = (int)(i % K);
        dst[i] = f2bf(src[(long long)k * N + n]);
    }
}

// ============ generic 32x64-tile GEMM body (double-buffered) ============
// Computes acc[2][4] for rows {m via arow0/arow1} x cols {n0..n0+63} over K=1024.
#define GEMM32x64_BODY(AROW0, AROW1, BPTR, BSTRIDE)                                   \
    const unsigned short* brow0 = (BPTR) + (long long)(n0 + 0 * 16 + nlo) * (BSTRIDE);\
    const unsigned short* brow1 = (BPTR) + (long long)(n0 + 1 * 16 + nlo) * (BSTRIDE);\
    const unsigned short* brow2 = (BPTR) + (long long)(n0 + 2 * 16 + nlo) * (BSTRIDE);\
    const unsigned short* brow3 = (BPTR) + (long long)(n0 + 3 * 16 + nlo) * (BSTRIDE);\
    v16bf a0 = ldfrag_bf16((AROW0), kh);                                              \
    v16bf a1 = ldfrag_bf16((AROW1), kh);                                              \
    v16bf b0 = ldfrag_bf16(brow0, kh);                                                \
    v16bf b1 = ldfrag_bf16(brow1, kh);                                                \
    v16bf b2 = ldfrag_bf16(brow2, kh);                                                \
    v16bf b3 = ldfrag_bf16(brow3, kh);                                                \
    for (int kc = 0; kc < D_MODEL; kc += 32) {                                        \
        int kn = (kc + 32 < D_MODEL) ? (kc + 32) : kc;                                \
        v16bf a0n = ldfrag_bf16((AROW0) + kn, kh);                                    \
        v16bf a1n = ldfrag_bf16((AROW1) + kn, kh);                                    \
        v16bf b0n = ldfrag_bf16(brow0 + kn, kh);                                      \
        v16bf b1n = ldfrag_bf16(brow1 + kn, kh);                                      \
        v16bf b2n = ldfrag_bf16(brow2 + kn, kh);                                      \
        v16bf b3n = ldfrag_bf16(brow3 + kn, kh);                                      \
        acc[0][0] = wmma_bf16(a0, b0, acc[0][0]);                                     \
        acc[1][0] = wmma_bf16(a1, b0, acc[1][0]);                                     \
        acc[0][1] = wmma_bf16(a0, b1, acc[0][1]);                                     \
        acc[1][1] = wmma_bf16(a1, b1, acc[1][1]);                                     \
        acc[0][2] = wmma_bf16(a0, b2, acc[0][2]);                                     \
        acc[1][2] = wmma_bf16(a1, b2, acc[1][2]);                                     \
        acc[0][3] = wmma_bf16(a0, b3, acc[0][3]);                                     \
        acc[1][3] = wmma_bf16(a1, b3, acc[1][3]);                                     \
        a0 = a0n; a1 = a1n; b0 = b0n; b1 = b1n; b2 = b2n; b3 = b3n;                   \
    }

// ---------------- pass 1a: Q = x @ wq  ->  Q[b][h][t][dh] bf16 ----------------
__global__ __launch_bounds__(256) void proj_q_kernel(const unsigned short* __restrict__ xbf,
                                                     const unsigned short* __restrict__ wqt,
                                                     unsigned short* __restrict__ Qd) {
    int lane = threadIdx.x & 31, widx = threadIdx.x >> 5;
    int w = blockIdx.x * 8 + widx;                 // 2048 waves total
    int nt = w & 15, mt = (w >> 4) & 31, b = w >> 9;
    int m0 = mt * 32, n0 = nt * 64;
    int nlo = lane & 15, hh = lane >> 4, kh = hh * 8;

    const unsigned short* arow0 = xbf + ((long long)(b * T_LEN + m0 + nlo)) * D_MODEL;
    const unsigned short* arow1 = arow0 + 16ll * D_MODEL;
    v8f acc[2][4] = {};
    GEMM32x64_BODY(arow0, arow1, wqt, D_MODEL)

#pragma unroll
    for (int mm = 0; mm < 2; mm++)
#pragma unroll
        for (int tt = 0; tt < 4; tt++) {
            int n = n0 + tt * 16 + nlo, h = n >> 6, dh = n & 63;
#pragma unroll
            for (int e = 0; e < 8; e++) {
                int i = m0 + mm * 16 + e + 8 * hh;
                Qd[(((long long)b * HEADS + h) * T_LEN + i) * DIM_H + dh] = f2bf(acc[mm][tt][e]);
            }
        }
}

// ---------------- pass 1b: [K|V] = concat(mem,x) @ wkv ----------------
// K -> Kd[b][h][s][dh] bf16 ; V -> Vtd[b][h][dh][s] bf16 (transposed for PV B-frags)
__global__ __launch_bounds__(256) void proj_kv_kernel(const unsigned short* __restrict__ xbf,
                                                      const unsigned short* __restrict__ membf,
                                                      const unsigned short* __restrict__ wkvt,
                                                      unsigned short* __restrict__ Kd,
                                                      unsigned short* __restrict__ Vtd) {
    int lane = threadIdx.x & 31, widx = threadIdx.x >> 5;
    int w = blockIdx.x * 8 + widx;                 // 8192 waves total
    int nt = w & 31, mt = (w >> 5) & 63, b = w >> 11;
    int m0 = mt * 32, n0 = nt * 64;
    int nlo = lane & 15, hh = lane >> 4, kh = hh * 8;

    int s0r = m0 + nlo, s1r = m0 + 16 + nlo;
    const unsigned short* arow0 = (s0r < MEM_LEN)
        ? (membf + ((long long)b * MEM_LEN + s0r) * D_MODEL)
        : (xbf   + ((long long)b * T_LEN + (s0r - MEM_LEN)) * D_MODEL);
    const unsigned short* arow1 = (s1r < MEM_LEN)
        ? (membf + ((long long)b * MEM_LEN + s1r) * D_MODEL)
        : (xbf   + ((long long)b * T_LEN + (s1r - MEM_LEN)) * D_MODEL);

    v8f acc[2][4] = {};
    GEMM32x64_BODY(arow0, arow1, wkvt, D_MODEL)

#pragma unroll
    for (int mm = 0; mm < 2; mm++)
#pragma unroll
        for (int tt = 0; tt < 4; tt++) {
            int n = n0 + tt * 16 + nlo;
#pragma unroll
            for (int e = 0; e < 8; e++) {
                int s = m0 + mm * 16 + e + 8 * hh;
                unsigned short val = f2bf(acc[mm][tt][e]);
                if (n < D_MODEL) {  // K half
                    int h = n >> 6, dh = n & 63;
                    Kd[(((long long)b * HEADS + h) * S_LEN + s) * DIM_H + dh] = val;
                } else {            // V half -> transposed
                    int nv = n - D_MODEL, h = nv >> 6, dh = nv & 63;
                    Vtd[(((long long)b * HEADS + h) * DIM_H + dh) * S_LEN + s] = val;
                }
            }
        }
}

// ---------------- pass 2: flash attention w/ rel-pos shift ----------------
// One wave = 16 query rows. Loops 32-key tiles with online softmax.
__global__ __launch_bounds__(256) void attn_kernel(const unsigned short* __restrict__ Q,
                                                   const unsigned short* __restrict__ K,
                                                   const unsigned short* __restrict__ Vt,
                                                   const unsigned short* __restrict__ relwbf,
                                                   unsigned short* __restrict__ AO) {
    __shared__ __align__(16) unsigned short ldsP[8 * 16 * 32];  // per-wave 16x32 bf16 P tile
    int lane = threadIdx.x & 31, widx = threadIdx.x >> 5;
    int w = blockIdx.x * 8 + widx;                 // 4096 waves total
    int qt = w & 63, bh = w >> 6, h = bh & 15;
    int qi0 = qt * 16;
    int nlo = lane & 15, hh = lane >> 4, kh = hh * 8;
    unsigned short* myP = ldsP + widx * 16 * 32;

    // Q A-fragments (kept in VGPRs for the whole loop)
    const unsigned short* qrow = Q + ((long long)bh * T_LEN + qi0 + nlo) * DIM_H;
    v16bf a0 = ldfrag_bf16(qrow, kh);        // k = 0..31
    v16bf a1 = ldfrag_bf16(qrow + 32, kh);   // k = 32..63

    const unsigned short* Krow = K  + (long long)bh * S_LEN * DIM_H;
    const unsigned short* Vrow = Vt + (long long)bh * DIM_H * S_LEN;
    const unsigned short* Wb = relwbf + (long long)h * DIM_H;   // row stride = D_MODEL

    v8f ot[4] = {v8f{}, v8f{}, v8f{}, v8f{}};
    float rmax[8], rsum[8];
#pragma unroll
    for (int e = 0; e < 8; e++) { rmax[e] = -3.0e38f; rsum[e] = 0.0f; }

    int limit = qi0 + 15 + MEM_LEN;                  // last valid key column
    int ntile = (limit >> 5) + 1;

    for (int it = 0; it < ntile; ++it) {
        int jj0 = it * 32;

        // prefetch next key tile (K rows + V rows) into cache while this tile computes
        {
            int jjn = min(jj0 + 32, S_LEN - 32);
            __builtin_prefetch(Krow + (long long)(jjn + nlo) * DIM_H, 0, 3);
            __builtin_prefetch(Krow + (long long)(jjn + 16 + nlo) * DIM_H, 0, 3);
            __builtin_prefetch(Vrow + (long long)(nlo) * S_LEN + jjn, 0, 3);
            __builtin_prefetch(Vrow + (long long)(48 + nlo) * S_LEN + jjn, 0, 3);
        }

        // ---- QK^T: 16x32 logits tile (2 n-subtiles x 2 k-chunks) ----
        const unsigned short* k0 = Krow + (long long)(jj0 + nlo) * DIM_H;
        const unsigned short* k1 = Krow + (long long)(jj0 + 16 + nlo) * DIM_H;
        v8f s0 = {}, s1 = {};
        s0 = wmma_bf16(a0, ldfrag_bf16(k0, kh), s0);
        s0 = wmma_bf16(a1, ldfrag_bf16(k0 + 32, kh), s0);
        s1 = wmma_bf16(a0, ldfrag_bf16(k1, kh), s1);
        s1 = wmma_bf16(a1, ldfrag_bf16(k1 + 32, kh), s1);

        // ---- rel-pos band: Praw[m][c] = q_m . w[jlo + c], c in [0,48) ----
        int jlo = jj0 + T_LEN - 16 - qi0;            // >= 0 always
        int ra = min(jlo + nlo,      S_LEN - 1);
        int rb = min(jlo + 16 + nlo, S_LEN - 1);
        int rc = min(jlo + 32 + nlo, S_LEN - 1);
        const unsigned short* p0 = Wb + (long long)ra * D_MODEL;
        const unsigned short* p1 = Wb + (long long)rb * D_MODEL;
        const unsigned short* p2 = Wb + (long long)rc * D_MODEL;
        v8f r0 = {}, r1 = {}, r2 = {};
        r0 = wmma_bf16(a0, ldfrag_bf16(p0, kh), r0);
        r0 = wmma_bf16(a1, ldfrag_bf16(p0 + 32, kh), r0);
        r1 = wmma_bf16(a0, ldfrag_bf16(p1, kh), r1);
        r1 = wmma_bf16(a1, ldfrag_bf16(p1 + 32, kh), r1);
        r2 = wmma_bf16(a0, ldfrag_bf16(p2, kh), r2);
        r2 = wmma_bf16(a1, ldfrag_bf16(p2 + 32, kh), r2);

        // ---- shift gather + mask + online softmax ----
#pragma unroll
        for (int e = 0; e < 8; e++) {
            int m = e + 8 * hh;
            int c = nlo + 15 - m;                    // [0,30]; source col for n-subtile 0
            int srcl = (c & 15) | (hh << 4);
            int bidx = srcl << 2;
            float v0 = __int_as_float(__builtin_amdgcn_ds_bpermute(bidx, __float_as_int(r0[e])));
            float v1 = __int_as_float(__builtin_amdgcn_ds_bpermute(bidx, __float_as_int(r1[e])));
            float v2 = __int_as_float(__builtin_amdgcn_ds_bpermute(bidx, __float_as_int(r2[e])));
            float pa = (c < 16) ? v0 : v1;           // c       in [0,30]
            float pb = (c < 16) ? v1 : v2;           // c+16    in [16,46]

            float l0 = (s0[e] + pa) * SCALE;
            float l1 = (s1[e] + pb) * SCALE;
            int i = qi0 + m;
            if (jj0 + nlo      > i + MEM_LEN) l0 = -1.0e30f;
            if (jj0 + 16 + nlo > i + MEM_LEN) l1 = -1.0e30f;

            float mx = fmaxf(l0, l1);
            mx = fmaxf(mx, __shfl_xor(mx, 1, 32));
            mx = fmaxf(mx, __shfl_xor(mx, 2, 32));
            mx = fmaxf(mx, __shfl_xor(mx, 4, 32));
            mx = fmaxf(mx, __shfl_xor(mx, 8, 32));
            float nm = fmaxf(rmax[e], mx);
            float corr = __expf(rmax[e] - nm);
            rmax[e] = nm;
            float e0 = __expf(l0 - nm);
            float e1 = __expf(l1 - nm);
            float ss = e0 + e1;
            ss += __shfl_xor(ss, 1, 32);
            ss += __shfl_xor(ss, 2, 32);
            ss += __shfl_xor(ss, 4, 32);
            ss += __shfl_xor(ss, 8, 32);
            rsum[e] = rsum[e] * corr + ss;
            ot[0][e] *= corr; ot[1][e] *= corr; ot[2][e] *= corr; ot[3][e] *= corr;
            myP[m * 32 + nlo]      = f2bf(e0);
            myP[m * 32 + 16 + nlo] = f2bf(e1);
        }

        // ---- P (C-layout) -> A-fragment via per-wave LDS repack ----
        v16bf pf = ldfrag_bf16(myP + nlo * 32, kh);  // A 16x32: lane m=nlo

        // ---- PV: O(16x64) += P(16x32) @ V(32x64) ----
#pragma unroll
        for (int tt = 0; tt < 4; tt++) {
            const unsigned short* vp = Vrow + (long long)(tt * 16 + nlo) * S_LEN + jj0;
            ot[tt] = wmma_bf16(pf, ldfrag_bf16(vp, kh), ot[tt]);
        }
    }

    // ---- normalize + store heads output bf16: AO[b*t][h*64+dh] ----
    float inv[8];
#pragma unroll
    for (int e = 0; e < 8; e++) inv[e] = 1.0f / rsum[e];
    long long aobase = ((long long)((bh >> 4) * T_LEN + qi0)) * D_MODEL + (long long)h * DIM_H;
#pragma unroll
    for (int tt = 0; tt < 4; tt++) {
#pragma unroll
        for (int e = 0; e < 8; e++) {
            int m = e + 8 * hh;
            AO[aobase + (long long)m * D_MODEL + tt * 16 + nlo] = f2bf(ot[tt][e] * inv[e]);
        }
    }
}

// ---------------- pass 3: out = AO @ wo + bo (fp32 out) ----------------
__global__ __launch_bounds__(256) void outproj_kernel(const unsigned short* __restrict__ AO,
                                                      const unsigned short* __restrict__ wot,
                                                      const float* __restrict__ bo,
                                                      float* __restrict__ out) {
    int lane = threadIdx.x & 31, widx = threadIdx.x >> 5;
    int w = blockIdx.x * 8 + widx;                 // 2048 waves total
    int nt = w & 15, mt = w >> 4;                  // mt in [0,128)
    int m0 = mt * 32, n0 = nt * 64;
    int nlo = lane & 15, hh = lane >> 4, kh = hh * 8;

    const unsigned short* arow0 = AO + (long long)(m0 + nlo) * D_MODEL;
    const unsigned short* arow1 = arow0 + 16ll * D_MODEL;
    v8f acc[2][4] = {};
    GEMM32x64_BODY(arow0, arow1, wot, D_MODEL)

#pragma unroll
    for (int mm = 0; mm < 2; mm++)
#pragma unroll
        for (int tt = 0; tt < 4; tt++) {
            int n = n0 + tt * 16 + nlo;
            float bias = bo[n];
#pragma unroll
            for (int e = 0; e < 8; e++) {
                int m = mm * 16 + e + 8 * hh;
                out[(long long)(m0 + m) * D_MODEL + n] = acc[mm][tt][e] + bias;
            }
        }
}

// ---------------- host-side orchestration ----------------
extern "C" void kernel_launch(void* const* d_in, const int* in_sizes, int n_in,
                              void* d_out, int out_size, void* d_ws, size_t ws_size,
                              hipStream_t stream) {
    const float* x    = (const float*)d_in[0];  // (4,1024,1024)
    const float* mem  = (const float*)d_in[1];  // (4,1024,1024)
    const float* wq   = (const float*)d_in[2];  // (1024,1024)
    const float* wkv  = (const float*)d_in[3];  // (1024,2048)
    const float* wo   = (const float*)d_in[4];  // (1024,1024)
    const float* bo   = (const float*)d_in[5];  // (1024,)
    const float* relw = (const float*)d_in[6];  // (2048,16,64)
    float* out = (float*)d_out;                 // out (4M floats) + mem_next (4M floats)

    // workspace layout (bf16 buffers), 76 MB total
    char* ws = (char*)d_ws;
    unsigned short* wqt    = (unsigned short*)(ws);                  // 2 MB  [1024][1024]
    unsigned short* wkvt   = (unsigned short*)(ws + (2ll  << 20));   // 4 MB  [2048][1024]
    unsigned short* wot    = (unsigned short*)(ws + (6ll  << 20));   // 2 MB  [1024][1024]
    unsigned short* Qd     = (unsigned short*)(ws + (8ll  << 20));   // 8 MB  [4][16][1024][64]
    unsigned short* Kd     = (unsigned short*)(ws + (16ll << 20));   // 16 MB [4][16][2048][64]
    unsigned short* Vtd    = (unsigned short*)(ws + (32ll << 20));   // 16 MB [4][16][64][2048]
    unsigned short* AO     = (unsigned short*)(ws + (48ll << 20));   // 8 MB  [4096][1024]
    unsigned short* xbf    = (unsigned short*)(ws + (56ll << 20));   // 8 MB  [4096][1024]
    unsigned short* membf  = (unsigned short*)(ws + (64ll << 20));   // 8 MB  [4096][1024]
    unsigned short* relwbf = (unsigned short*)(ws + (72ll << 20));   // 4 MB  [2048][1024]

    // pass 0: activations/rel_w -> bf16; weights -> bf16 transposed [out][in]
    convert_bf16_kernel<<<(4 * 1024 * 1024) / 256, 256, 0, stream>>>(x,    xbf,    4ll * 1024 * 1024);
    convert_bf16_kernel<<<(4 * 1024 * 1024) / 256, 256, 0, stream>>>(mem,  membf,  4ll * 1024 * 1024);
    convert_bf16_kernel<<<(2 * 1024 * 1024) / 256, 256, 0, stream>>>(relw, relwbf, 2ll * 1024 * 1024);
    transpose_bf16_kernel<<<(1024 * 1024) / 256, 256, 0, stream>>>(wq,  wqt,  1024, 1024);
    transpose_bf16_kernel<<<(1024 * 2048) / 256, 256, 0, stream>>>(wkv, wkvt, 1024, 2048);
    transpose_bf16_kernel<<<(1024 * 1024) / 256, 256, 0, stream>>>(wo,  wot,  1024, 1024);

    // pass 1: projections (32x64 tiles, double-buffered)
    proj_q_kernel<<<256, 256, 0, stream>>>(xbf, wqt, Qd);
    proj_kv_kernel<<<1024, 256, 0, stream>>>(xbf, membf, wkvt, Kd, Vtd);

    // pass 2: attention
    attn_kernel<<<512, 256, 0, stream>>>(Qd, Kd, Vtd, relwbf, AO);

    // pass 3: output projection
    outproj_kernel<<<256, 256, 0, stream>>>(AO, wot, bo, out);

    // mem_next = concat(mem, x)[:, -1024:] == x  (mem_len == t)
    hipMemcpyAsync(out + (4ll * 1024 * 1024), x, 4ll * 1024 * 1024 * sizeof(float),
                   hipMemcpyDeviceToDevice, stream);
}
